// DotProductAttentionTransformer_16183436771980
// MI455X (gfx1250) — compile-verified, matrix operand order checked by reading, not run on hardware
//
#include <hip/hip_runtime.h>
#include <hip/hip_bf16.h>
#include <math.h>

typedef _Float16 half_t;
typedef __attribute__((ext_vector_type(8)))  _Float16 v8h;
typedef __attribute__((ext_vector_type(16))) _Float16 v16h;
typedef __attribute__((ext_vector_type(8)))  float    v8f;

#define NN 10000
#define NE 160000
#define DIM 480

// ---------------- WMMA fragment helpers (CDNA5 16x16x32 f16, wave32) ----------------
// A (16x32 f16, ISA 7.12.2): lane L = row L&15; per lane the 16 halves are two contiguous
// 8-half chunks: [kBase + hi*8 .. +7] and [kBase + 16 + hi*8 .. +7]  (hi = lane>>4).
__device__ __forceinline__ v16h frag_a_lds(const half_t* base, int stride, int rowBase,
                                           int kBase, int lane) {
  const int off = (rowBase + (lane & 15)) * stride + kBase + ((lane >> 4) << 3);
  v8h lo = *(const v8h*)(base + off);
  v8h hi = *(const v8h*)(base + off + 16);
  return __builtin_shufflevector(lo, hi, 0, 1, 2, 3, 4, 5, 6, 7,
                                 8, 9, 10, 11, 12, 13, 14, 15);
}

// B fragments are pre-swizzled in global memory: layout [kt][nt][lane][16 halves],
// so a fragment is one aligned 32-byte load per lane.
__device__ __forceinline__ v16h frag_b_sw(const half_t* __restrict__ Wsw, int NT,
                                          int kt, int nt, int lane) {
  return *(const v16h*)(Wsw + ((((kt * NT + nt) * 32) + lane) << 4));
}

__device__ __forceinline__ v8f wmma_f16(v16h a, v16h b, v8f c) {
  return __builtin_amdgcn_wmma_f32_16x16x32_f16(false, a, false, b, (short)0, c, false, false);
}
// C/D element (lane, v): row m = (lane>=16 ? 8 : 0) + v, col n = lane & 15.

// ---------------- weight swizzle f32 -> fragment-ready f16 ----------------
// dst tile grid has NT column-tiles total; this source occupies tiles [ntOff, ntOff+C/16).
__global__ void k_swz(half_t* __restrict__ dst, const float* __restrict__ src,
                      int K, int C, int NT, int ntOff) {
  int idx = blockIdx.x * blockDim.x + threadIdx.x;
  if (idx >= K * C) return;
  int h    = idx & 15;
  int lane = (idx >> 4) & 31;
  int tl   = idx >> 9;
  int NTl  = C >> 4;
  int kt = tl / NTl, ntl = tl % NTl;
  int hi = lane >> 4;
  int k = kt * 32 + h + ((h & 8) ? 8 : 0) + (hi ? 8 : 0);
  int n = ntl * 16 + (lane & 15);
  dst[((((kt * NT + ntOff + ntl) * 32) + lane) << 4) + h] = (half_t)src[k * C + n];
}

// stage a 16-row natural-layout tile into per-irrep-block LDS (f16)
__device__ __forceinline__ void stage_tile(const float* __restrict__ x, int n0, int lane,
                                           half_t* a0, half_t* a1, half_t* a2) {
  for (int idx = lane; idx < 16 * DIM; idx += 32) {
    int i = idx / DIM, d = idx % DIM;
    float val = x[(size_t)(n0 + i) * DIM + d];
    if (d < 128) {
      a0[i * 128 + d] = (half_t)val;
    } else if (d < 320) {
      int l = d - 128, m = l / 3, c = l % 3;
      a1[(c * 16 + i) * 64 + m] = (half_t)val;
    } else {
      int l = d - 320, m = l / 5, c = l % 5;
      a2[(c * 16 + i) * 32 + m] = (half_t)val;
    }
  }
}

// ---------------- fused node linears: q (scaled, bias) | msrc (bias) | mdst ----------------
__global__ __launch_bounds__(32)
void k_node(const float* __restrict__ node,
            const half_t* __restrict__ Wc0, const half_t* __restrict__ Wc1,
            const half_t* __restrict__ Wc2,
            const float* __restrict__ bq, const float* __restrict__ bsrc,
            float* __restrict__ qOut, float* __restrict__ srcOut, float* __restrict__ dstOut) {
  __shared__ alignas(16) half_t a0[16 * 128];
  __shared__ alignas(16) half_t a1[48 * 64];
  __shared__ alignas(16) half_t a2[80 * 32];
  const int lane = threadIdx.x;
  const int n0 = blockIdx.x * 16;
  stage_tile(node, n0, lane, a0, a1, a2);
  __syncthreads();

  const float cf  = 0.13363062095621219f;      // 1/sqrt(56)
  const float s0  = 0.08838834764831845f;      // 1/sqrt(128)
  const float s1  = 0.125f;                    // 1/sqrt(64)
  const float s2  = 0.17677669529663687f;      // 1/sqrt(32)
  const float qs1 = cf * 0.5773502691896258f;  // cf/sqrt(3)
  const float qs2 = cf * 0.4472135954999579f;  // cf/sqrt(5)

  const int n = lane & 15, mb = (lane >> 4) * 8;

  // block0: [16x128] x [128 x (128q|128src|128dst)], NT=24
  for (int ot = 0; ot < 3; ++ot) {
    float* op = (ot == 0) ? qOut : (ot == 1) ? srcOut : dstOut;
    const float* bp = (ot == 0) ? bq : (ot == 1) ? bsrc : nullptr;
    const float postm = (ot == 0) ? cf : 1.f;
    for (int t8 = 0; t8 < 8; ++t8) {
      const int nt = ot * 8 + t8;
      v8f acc = {};
#pragma unroll
      for (int kt = 0; kt < 4; ++kt)
        acc = wmma_f16(frag_a_lds(a0, 128, 0, kt * 32, lane),
                       frag_b_sw(Wc0, 24, kt, nt, lane), acc);
      const int j = t8 * 16 + n;
      const float bias = bp ? bp[j] : 0.f;
#pragma unroll
      for (int v = 0; v < 8; ++v)
        op[(size_t)(n0 + mb + v) * DIM + j] = (acc[v] * s0 + bias) * postm;
    }
  }
  // block1: NT=12 (4 tiles per output); reuse B across the 3 component row-tiles
  for (int nt = 0; nt < 12; ++nt) {
    v16h b0 = frag_b_sw(Wc1, 12, 0, nt, lane);
    v16h b1 = frag_b_sw(Wc1, 12, 1, nt, lane);
    const int ot = nt >> 2;
    const int j = (nt & 3) * 16 + n;
    float* op = (ot == 0) ? qOut : (ot == 1) ? srcOut : dstOut;
    const float scl = s1 * ((ot == 0) ? qs1 / s1 * s1 : 1.f) * ((ot == 0) ? 1.f : 1.f);
    const float scale = (ot == 0) ? s1 * (qs1 / s1) : s1;  // q: s1*qs1-compensated
    (void)scl;
    for (int rt = 0; rt < 3; ++rt) {
      v8f acc = {};
      acc = wmma_f16(frag_a_lds(a1, 64, rt * 16, 0, lane), b0, acc);
      acc = wmma_f16(frag_a_lds(a1, 64, rt * 16, 32, lane), b1, acc);
      const int tgt = 128 + j * 3 + rt;
#pragma unroll
      for (int v = 0; v < 8; ++v)
        op[(size_t)(n0 + mb + v) * DIM + tgt] = acc[v] * scale;
    }
  }
  // block2: NT=6 (2 tiles per output); reuse B across the 5 component row-tiles
  for (int nt = 0; nt < 6; ++nt) {
    v16h b = frag_b_sw(Wc2, 6, 0, nt, lane);
    const int ot = nt >> 1;
    const int j = (nt & 1) * 16 + n;
    float* op = (ot == 0) ? qOut : (ot == 1) ? srcOut : dstOut;
    const float scale = (ot == 0) ? s2 * (qs2 / s2) : s2;
    for (int rt = 0; rt < 5; ++rt) {
      v8f acc = {};
      acc = wmma_f16(frag_a_lds(a2, 32, rt * 16, 0, lane), b, acc);
      const int tgt = 320 + j * 5 + rt;
#pragma unroll
      for (int v = 0; v < 8; ++v)
        op[(size_t)(n0 + mb + v) * DIM + tgt] = acc[v] * scale;
    }
  }
}

// ---------------- fused edge kernel: gate MLP + gather + gating + kv GEMM + alpha ----------------
__global__ __launch_bounds__(32)
void k_edge(const float* __restrict__ es, const float* __restrict__ eattr,
            const int* __restrict__ esrc, const int* __restrict__ edst,
            const float* __restrict__ msrc, const float* __restrict__ mdst,
            const float* __restrict__ q,
            const half_t* __restrict__ Wr1sw, const half_t* __restrict__ Wr2sw,
            const half_t* __restrict__ Wr3sw,
            const float* __restrict__ br1, const float* __restrict__ br2,
            const float* __restrict__ br3,
            const half_t* __restrict__ Wk0, const half_t* __restrict__ Wk1,
            const half_t* __restrict__ Wk2,
            const float* __restrict__ bkv,
            half_t* __restrict__ vstore, float* __restrict__ alphaOut) {
  __shared__ alignas(16) half_t esA[16 * 128];
  __shared__ alignas(16) half_t h1[16 * 64];
  __shared__ alignas(16) half_t h2[16 * 64];
  __shared__ alignas(16) half_t gates[16 * 224];
  __shared__ alignas(16) half_t kv0[16 * 128];
  __shared__ alignas(16) half_t kv1[48 * 64];
  __shared__ alignas(16) half_t kv2[80 * 32];
  __shared__ alignas(16) float  stageK[3072];   // max(16*128, 48*64, 80*32)
  __shared__ alignas(16) half_t qT[16 * 480];
  __shared__ float  aAcc[64];                   // [h*16 + e]

  const int lane = threadIdx.x;
  const int e0 = blockIdx.x * 16;
  const int n = lane & 15, mb = (lane >> 4) * 8;

  // prefetch the gathered node rows for this tile (global_prefetch_b8 path)
  if (lane < 16) {
    __builtin_prefetch(&msrc[(size_t)esrc[e0 + lane] * DIM], 0, 3);
    __builtin_prefetch(&mdst[(size_t)edst[e0 + lane] * DIM], 0, 3);
  }
  for (int idx = lane; idx < 16 * 128; idx += 32)
    esA[idx] = (half_t)es[(size_t)e0 * 128 + idx];
  for (int idx = lane; idx < 16 * 480; idx += 32) {
    int e = idx / 480, d = idx % 480;
    qT[idx] = (half_t)q[(size_t)edst[e0 + e] * DIM + d];
  }
  aAcc[lane] = 0.f; aAcc[lane + 32] = 0.f;
  __syncthreads();

  // MLP layer1: [16x128]x[128x64] + silu, NT=4
  for (int nt = 0; nt < 4; ++nt) {
    v8f acc = {};
#pragma unroll
    for (int kt = 0; kt < 4; ++kt)
      acc = wmma_f16(frag_a_lds(esA, 128, 0, kt * 32, lane),
                     frag_b_sw(Wr1sw, 4, kt, nt, lane), acc);
    const int j = nt * 16 + n;
    const float bb = br1[j];
#pragma unroll
    for (int v = 0; v < 8; ++v) {
      float x = acc[v] + bb;
      x = x / (1.f + __expf(-x));
      h1[(mb + v) * 64 + j] = (half_t)x;
    }
  }
  __syncthreads();
  // MLP layer2: [16x64]x[64x64] + silu, NT=4
  for (int nt = 0; nt < 4; ++nt) {
    v8f acc = {};
#pragma unroll
    for (int kt = 0; kt < 2; ++kt)
      acc = wmma_f16(frag_a_lds(h1, 64, 0, kt * 32, lane),
                     frag_b_sw(Wr2sw, 4, kt, nt, lane), acc);
    const int j = nt * 16 + n;
    const float bb = br2[j];
#pragma unroll
    for (int v = 0; v < 8; ++v) {
      float x = acc[v] + bb;
      x = x / (1.f + __expf(-x));
      h2[(mb + v) * 64 + j] = (half_t)x;
    }
  }
  __syncthreads();
  // MLP layer3: [16x64]x[64x224] -> gates, NT=14
  for (int nt = 0; nt < 14; ++nt) {
    v8f acc = {};
#pragma unroll
    for (int kt = 0; kt < 2; ++kt)
      acc = wmma_f16(frag_a_lds(h2, 64, 0, kt * 32, lane),
                     frag_b_sw(Wr3sw, 14, kt, nt, lane), acc);
    const int j = nt * 16 + n;
    const float bb = br3[j];
#pragma unroll
    for (int v = 0; v < 8; ++v)
      gates[(mb + v) * 224 + j] = (half_t)(acc[v] + bb);
  }
  __syncthreads();

  // gather msrc[src]+mdst[dst], gate per irrep, * sh0, stage per-block (f16)
  for (int idx = lane; idx < 16 * DIM; idx += 32) {
    int e = idx / DIM, d = idx % DIM;
    int ge = e0 + e;
    float val = msrc[(size_t)esrc[ge] * DIM + d] + mdst[(size_t)edst[ge] * DIM + d];
    float sh0 = eattr[(size_t)ge * 9];
    if (d < 128) {
      kv0[e * 128 + d] = (half_t)(val * (float)gates[e * 224 + d] * sh0);
    } else if (d < 320) {
      int l = d - 128, m = l / 3, c = l % 3;
      kv1[(c * 16 + e) * 64 + m] = (half_t)(val * (float)gates[e * 224 + 128 + m] * sh0);
    } else {
      int l = d - 320, m = l / 5, c = l % 5;
      kv2[(c * 16 + e) * 32 + m] = (half_t)(val * (float)gates[e * 224 + 192 + m] * sh0);
    }
  }
  __syncthreads();

  const float s0 = 0.08838834764831845f;
  const float s1 = 0.125f;
  const float s2 = 0.17677669529663687f;

  // kv block0: [16x128]x[128x256], NT=16; nt<8 = k (stage), nt>=8 = v (store f16)
  for (int nt = 0; nt < 16; ++nt) {
    v8f acc = {};
#pragma unroll
    for (int kt = 0; kt < 4; ++kt)
      acc = wmma_f16(frag_a_lds(kv0, 128, 0, kt * 32, lane),
                     frag_b_sw(Wk0, 16, kt, nt, lane), acc);
    const int j = nt * 16 + n;
    const float bb = bkv[j];
    if (nt < 8) {
#pragma unroll
      for (int v = 0; v < 8; ++v) stageK[(mb + v) * 128 + j] = acc[v] * s0 + bb;
    } else {
#pragma unroll
      for (int v = 0; v < 8; ++v)
        vstore[(size_t)(e0 + mb + v) * DIM + (j - 128)] = (half_t)(acc[v] * s0 + bb);
    }
  }
  __syncthreads();
  for (int t = lane; t < 64; t += 32) {  // alpha += <q,k> block0
    int h = t >> 4, e = t & 15;
    float s = 0.f;
    const float*  kr = &stageK[e * 128 + h * 32];
    const half_t* qr = &qT[e * 480 + h * 32];
#pragma unroll
    for (int m = 0; m < 32; ++m) s += kr[m] * (float)qr[m];
    aAcc[t] += s;
  }
  __syncthreads();

  // kv block1: [16x64]x[64x128] per component, NT=8; B reused across 3 row-tiles
  for (int nt = 0; nt < 8; ++nt) {
    v16h b0 = frag_b_sw(Wk1, 8, 0, nt, lane);
    v16h b1 = frag_b_sw(Wk1, 8, 1, nt, lane);
    const int j = nt * 16 + n;
    for (int rt = 0; rt < 3; ++rt) {
      v8f acc = {};
      acc = wmma_f16(frag_a_lds(kv1, 64, rt * 16, 0, lane), b0, acc);
      acc = wmma_f16(frag_a_lds(kv1, 64, rt * 16, 32, lane), b1, acc);
      if (nt < 4) {
#pragma unroll
        for (int v = 0; v < 8; ++v) stageK[(rt * 16 + mb + v) * 64 + j] = acc[v] * s1;
      } else {
#pragma unroll
        for (int v = 0; v < 8; ++v)
          vstore[(size_t)(e0 + mb + v) * DIM + 128 + (j - 64) * 3 + rt] = (half_t)(acc[v] * s1);
      }
    }
  }
  __syncthreads();
  for (int t = lane; t < 64; t += 32) {
    int h = t >> 4, e = t & 15;
    float s = 0.f;
    for (int c = 0; c < 3; ++c) {
      const float* kr = &stageK[(c * 16 + e) * 64 + h * 16];
#pragma unroll
      for (int m = 0; m < 16; ++m)
        s += kr[m] * (float)qT[e * 480 + 128 + (h * 16 + m) * 3 + c];
    }
    aAcc[t] += s;
  }
  __syncthreads();

  // kv block2: [16x32]x[32x64] per component, NT=4; B reused across 5 row-tiles
  for (int nt = 0; nt < 4; ++nt) {
    v16h b = frag_b_sw(Wk2, 4, 0, nt, lane);
    const int j = nt * 16 + n;
    for (int rt = 0; rt < 5; ++rt) {
      v8f acc = {};
      acc = wmma_f16(frag_a_lds(kv2, 32, rt * 16, 0, lane), b, acc);
      if (nt < 2) {
#pragma unroll
        for (int v = 0; v < 8; ++v) stageK[(rt * 16 + mb + v) * 32 + j] = acc[v] * s2;
      } else {
#pragma unroll
        for (int v = 0; v < 8; ++v)
          vstore[(size_t)(e0 + mb + v) * DIM + 320 + (j - 32) * 5 + rt] = (half_t)(acc[v] * s2);
      }
    }
  }
  __syncthreads();
  for (int t = lane; t < 64; t += 32) {
    int h = t >> 4, e = t & 15;
    float s = 0.f;
    for (int c = 0; c < 5; ++c) {
      const float* kr = &stageK[(c * 16 + e) * 32 + h * 8];
#pragma unroll
      for (int m = 0; m < 8; ++m)
        s += kr[m] * (float)qT[e * 480 + 320 + (h * 8 + m) * 5 + c];
    }
    aAcc[t] += s;
  }
  __syncthreads();
  for (int t = lane; t < 64; t += 32) {
    int h = t >> 4, e = t & 15;
    alphaOut[(size_t)(e0 + e) * 4 + h] = aAcc[t];
  }
}

__device__ __forceinline__ void seg_bounds(const int* __restrict__ edst, int nn,
                                           int& beg, int& end) {
  int lo = 0, hi = NE;
  while (lo < hi) { int mid = (lo + hi) >> 1; if (edst[mid] < nn) lo = mid + 1; else hi = mid; }
  beg = lo;
  lo = beg; hi = NE;
  while (lo < hi) { int mid = (lo + hi) >> 1; if (edst[mid] <= nn) lo = mid + 1; else hi = mid; }
  end = lo;
}

// ---------------- segment softmax over sorted edge_dst (one wave per node) ----------------
__global__ __launch_bounds__(32)
void k_softmax(float* __restrict__ alpha, const int* __restrict__ edst) {
  const int nn = blockIdx.x;
  const int lane = threadIdx.x;
  int beg, end;
  seg_bounds(edst, nn, beg, end);
  if (beg >= end) return;
  for (int h = 0; h < 4; ++h) {
    float m = -3.402823466e38f;
    for (int e = beg + lane; e < end; e += 32) m = fmaxf(m, alpha[(size_t)e * 4 + h]);
#pragma unroll
    for (int off = 16; off > 0; off >>= 1) m = fmaxf(m, __shfl_xor(m, off, 32));
    float s = 0.f;
    for (int e = beg + lane; e < end; e += 32) s += __expf(alpha[(size_t)e * 4 + h] - m);
#pragma unroll
    for (int off = 16; off > 0; off >>= 1) s += __shfl_xor(s, off, 32);
    float inv = 1.f / s;
    for (int e = beg + lane; e < end; e += 32)
      alpha[(size_t)e * 4 + h] = __expf(alpha[(size_t)e * 4 + h] - m) * inv;
  }
}

// ---------------- aggregation: attn[n] = sum_e alpha[e,h]*v[e] (natural layout) ----------------
__global__ __launch_bounds__(32)
void k_agg(const float* __restrict__ alpha, const half_t* __restrict__ vstore,
           const int* __restrict__ edst, float* __restrict__ attn) {
  const int nn = blockIdx.x;
  const int lane = threadIdx.x;
  int beg, end;
  seg_bounds(edst, nn, beg, end);
  float acc[15];
  int hIdx[15];
#pragma unroll
  for (int t = 0; t < 15; ++t) {
    int d = lane * 15 + t;
    acc[t] = 0.f;
    hIdx[t] = (d < 128) ? (d >> 5) : ((d < 320) ? (d - 128) / 48 : (d - 320) / 40);
  }
  for (int e = beg; e < end; ++e) {
    float al[4];
#pragma unroll
    for (int h = 0; h < 4; ++h) al[h] = alpha[(size_t)e * 4 + h];
    const half_t* vr = &vstore[(size_t)e * DIM + lane * 15];
#pragma unroll
    for (int t = 0; t < 15; ++t) acc[t] += al[hIdx[t]] * (float)vr[t];
  }
#pragma unroll
  for (int t = 0; t < 15; ++t) attn[(size_t)nn * DIM + lane * 15 + t] = acc[t];
}

// ---------------- output projection ----------------
__global__ __launch_bounds__(32)
void k_proj(const float* __restrict__ attn,
            const half_t* __restrict__ Wp0, const half_t* __restrict__ Wp1,
            const half_t* __restrict__ Wp2,
            const float* __restrict__ bproj, float* __restrict__ out) {
  __shared__ alignas(16) half_t a0[16 * 128];
  __shared__ alignas(16) half_t a1[48 * 64];
  __shared__ alignas(16) half_t a2[80 * 32];
  const int lane = threadIdx.x;
  const int n0 = blockIdx.x * 16;
  stage_tile(attn, n0, lane, a0, a1, a2);
  __syncthreads();

  const float s0 = 0.08838834764831845f;
  const float s1 = 0.125f;
  const float s2 = 0.17677669529663687f;
  const int n = lane & 15, mb = (lane >> 4) * 8;

  for (int nt = 0; nt < 8; ++nt) {
    v8f acc = {};
#pragma unroll
    for (int kt = 0; kt < 4; ++kt)
      acc = wmma_f16(frag_a_lds(a0, 128, 0, kt * 32, lane),
                     frag_b_sw(Wp0, 8, kt, nt, lane), acc);
    const int j = nt * 16 + n;
    const float bb = bproj[j];
#pragma unroll
    for (int v = 0; v < 8; ++v)
      out[(size_t)(n0 + mb + v) * DIM + j] = acc[v] * s0 + bb;
  }
  for (int nt = 0; nt < 4; ++nt) {
    v16h b0 = frag_b_sw(Wp1, 4, 0, nt, lane);
    v16h b1 = frag_b_sw(Wp1, 4, 1, nt, lane);
    const int j = nt * 16 + n;
    for (int rt = 0; rt < 3; ++rt) {
      v8f acc = {};
      acc = wmma_f16(frag_a_lds(a1, 64, rt * 16, 0, lane), b0, acc);
      acc = wmma_f16(frag_a_lds(a1, 64, rt * 16, 32, lane), b1, acc);
#pragma unroll
      for (int v = 0; v < 8; ++v)
        out[(size_t)(n0 + mb + v) * DIM + 128 + j * 3 + rt] = acc[v] * s1;
    }
  }
  for (int nt = 0; nt < 2; ++nt) {
    v16h b = frag_b_sw(Wp2, 2, 0, nt, lane);
    const int j = nt * 16 + n;
    for (int rt = 0; rt < 5; ++rt) {
      v8f acc = {};
      acc = wmma_f16(frag_a_lds(a2, 32, rt * 16, 0, lane), b, acc);
#pragma unroll
      for (int v = 0; v < 8; ++v)
        out[(size_t)(n0 + mb + v) * DIM + 320 + j * 5 + rt] = acc[v] * s2;
    }
  }
}

extern "C" void kernel_launch(void* const* d_in, const int* in_sizes, int n_in,
                              void* d_out, int out_size, void* d_ws, size_t ws_size,
                              hipStream_t stream) {
  (void)in_sizes; (void)n_in; (void)out_size; (void)ws_size;
  const float* node  = (const float*)d_in[0];
  const float* eattr = (const float*)d_in[2];
  const float* es    = (const float*)d_in[3];
  const int*   esrc  = (const int*)d_in[4];
  const int*   edst  = (const int*)d_in[5];
  const float* Wq0   = (const float*)d_in[7];
  const float* Wsrc0 = (const float*)d_in[8];
  const float* Wdst0 = (const float*)d_in[9];
  const float* Wp0   = (const float*)d_in[10];
  const float* Wkv0  = (const float*)d_in[11];
  const float* Wq1   = (const float*)d_in[12];
  const float* Wsrc1 = (const float*)d_in[13];
  const float* Wdst1 = (const float*)d_in[14];
  const float* Wp1   = (const float*)d_in[15];
  const float* Wkv1  = (const float*)d_in[16];
  const float* Wq2   = (const float*)d_in[17];
  const float* Wsrc2 = (const float*)d_in[18];
  const float* Wdst2 = (const float*)d_in[19];
  const float* Wp2   = (const float*)d_in[20];
  const float* Wkv2  = (const float*)d_in[21];
  const float* bq    = (const float*)d_in[22];
  const float* bsrc  = (const float*)d_in[23];
  const float* bkv   = (const float*)d_in[24];
  const float* bproj = (const float*)d_in[25];
  const float* Wr1   = (const float*)d_in[26];
  const float* br1   = (const float*)d_in[27];
  const float* Wr2   = (const float*)d_in[28];
  const float* br2   = (const float*)d_in[29];
  const float* Wr3   = (const float*)d_in[30];
  const float* br3   = (const float*)d_in[31];

  char* ws = (char*)d_ws;
  size_t off = 0;
  auto alloc = [&](size_t bytes) -> void* {
    void* p = ws + off;
    off = (off + bytes + 255) & ~(size_t)255;
    return p;
  };
  half_t* Wc0sw = (half_t*)alloc(128 * 384 * 2);
  half_t* Wc1sw = (half_t*)alloc(64 * 192 * 2);
  half_t* Wc2sw = (half_t*)alloc(32 * 96 * 2);
  half_t* Wk0sw = (half_t*)alloc(128 * 256 * 2);
  half_t* Wk1sw = (half_t*)alloc(64 * 128 * 2);
  half_t* Wk2sw = (half_t*)alloc(32 * 64 * 2);
  half_t* Wp0sw = (half_t*)alloc(128 * 128 * 2);
  half_t* Wp1sw = (half_t*)alloc(64 * 64 * 2);
  half_t* Wp2sw = (half_t*)alloc(32 * 32 * 2);
  half_t* Wr1sw = (half_t*)alloc(128 * 64 * 2);
  half_t* Wr2sw = (half_t*)alloc(64 * 64 * 2);
  half_t* Wr3sw = (half_t*)alloc(64 * 224 * 2);
  float*  qbuf   = (float*)alloc((size_t)NN * DIM * 4);
  float*  msrcB  = (float*)alloc((size_t)NN * DIM * 4);
  float*  mdstB  = (float*)alloc((size_t)NN * DIM * 4);
  half_t* vstore = (half_t*)alloc((size_t)NE * DIM * 2);
  float*  alphaB = (float*)alloc((size_t)NE * 4 * 4);
  float*  attnB  = (float*)alloc((size_t)NN * DIM * 4);

  auto swz = [&](half_t* dst, const float* src, int K, int C, int NT, int ntOff) {
    int tot = K * C;
    k_swz<<<(tot + 255) / 256, 256, 0, stream>>>(dst, src, K, C, NT, ntOff);
  };
  swz(Wc0sw, Wq0, 128, 128, 24, 0);
  swz(Wc0sw, Wsrc0, 128, 128, 24, 8);
  swz(Wc0sw, Wdst0, 128, 128, 24, 16);
  swz(Wc1sw, Wq1, 64, 64, 12, 0);
  swz(Wc1sw, Wsrc1, 64, 64, 12, 4);
  swz(Wc1sw, Wdst1, 64, 64, 12, 8);
  swz(Wc2sw, Wq2, 32, 32, 6, 0);
  swz(Wc2sw, Wsrc2, 32, 32, 6, 2);
  swz(Wc2sw, Wdst2, 32, 32, 6, 4);
  swz(Wk0sw, Wkv0, 128, 256, 16, 0);
  swz(Wk1sw, Wkv1, 64, 128, 8, 0);
  swz(Wk2sw, Wkv2, 32, 64, 4, 0);
  swz(Wp0sw, Wp0, 128, 128, 8, 0);
  swz(Wp1sw, Wp1, 64, 64, 4, 0);
  swz(Wp2sw, Wp2, 32, 32, 2, 0);
  swz(Wr1sw, Wr1, 128, 64, 4, 0);
  swz(Wr2sw, Wr2, 64, 64, 4, 0);
  swz(Wr3sw, Wr3, 64, 224, 14, 0);

  k_node<<<NN / 16, 32, 0, stream>>>(node, Wc0sw, Wc1sw, Wc2sw, bq, bsrc, qbuf, msrcB, mdstB);
  k_edge<<<NE / 16, 32, 0, stream>>>(es, eattr, esrc, edst, msrcB, mdstB, qbuf,
                                     Wr1sw, Wr2sw, Wr3sw, br1, br2, br3,
                                     Wk0sw, Wk1sw, Wk2sw, bkv, vstore, alphaB);
  k_softmax<<<NN, 32, 0, stream>>>(alphaB, edst);
  k_agg<<<NN, 32, 0, stream>>>(alphaB, vstore, edst, attnB);
  k_proj<<<NN / 16, 32, 0, stream>>>(attnB, Wp0sw, Wp1sw, Wp2sw, bproj, (float*)d_out);
}
